// OptimizedQuantumAI_61375082660343
// MI455X (gfx1250) — matrix operand order repaired and checked by reading, not compiled, not run on hardware
//
#include <hip/hip_runtime.h>
#include <hip/hip_bf16.h>
#include <math.h>

// ---------------------------------------------------------------------------
// QuantumAI pipeline on gfx1250 (wave32, WMMA).
//   encoder (39->64->32->4) on v_wmma_f32_16x16x32_f16,
//   4-qubit statevector sim + decoder (4->32->16->8) per-lane on VALU.
// B = 262144, 256 samples per 256-thread block => grid 1024.
// ---------------------------------------------------------------------------

typedef __attribute__((ext_vector_type(16))) _Float16 v16h;
typedef __attribute__((ext_vector_type(8)))  _Float16 v8h;
typedef __attribute__((ext_vector_type(8)))  float    v8f;

#define PI_F 3.14159265358979323846f

static __device__ inline v8f wmma_f16(v16h a, v16h b, v8f c) {
  // (neg_a, A, neg_b, B, c_mod, C, reuse_a, reuse_b)
  return __builtin_amdgcn_wmma_f32_16x16x32_f16(false, a, false, b,
                                                (short)0, c, false, false);
}

// A-fragment (16x32 f16): lane holds row (lane&15); 16 contiguous K:
// K = k0..k0+7 (VGPR0..3) and K = k0+8..k0+15 (VGPR4..7),
// where k0 = ks*32 + (lane>>4)*16.
static __device__ inline v16h load_a_frag(const _Float16* rowp, int k0) {
  v8h lo = *(const v8h*)(rowp + k0);
  v8h hi = *(const v8h*)(rowp + k0 + 8);
  return __builtin_shufflevector(lo, hi, 0, 1, 2, 3, 4, 5, 6, 7,
                                 8, 9, 10, 11, 12, 13, 14, 15);
}

__global__ __launch_bounds__(256) void qai_fused_kernel(
    const float* __restrict__ state,
    const float* __restrict__ enc_w1, const float* __restrict__ enc_b1,
    const float* __restrict__ enc_w2, const float* __restrict__ enc_b2,
    const float* __restrict__ enc_w3, const float* __restrict__ enc_b3,
    const float* __restrict__ qparams,
    const float* __restrict__ dec_w1, const float* __restrict__ dec_b1,
    const float* __restrict__ dec_w2, const float* __restrict__ dec_b2,
    const float* __restrict__ dec_w3, const float* __restrict__ dec_b3,
    float* __restrict__ out)
{
  // LDS: weights (f16, padded) + per-wave activation tiles + encoded values.
  __shared__ __align__(32) _Float16 sW1[64 * 64];   // [n=64][k=64], k 39..63 = 0
  __shared__ __align__(32) _Float16 sW2[32 * 64];   // [n=32][k=64]
  __shared__ __align__(32) _Float16 sW3[16 * 32];   // [n=16][k=32], n 4..15 = 0
  __shared__ __align__(32) _Float16 sX[8][32 * 64]; // per-wave [row=32][k=64]
  __shared__ float sEnc[8][32 * 4];                 // per-wave encoded [row][4]

  const int tid   = threadIdx.x;
  const int wave  = tid >> 5;
  const int lane  = tid & 31;
  const int laneN = lane & 15;  // N column / A row selector
  const int laneH = lane >> 4;  // which K-half this lane carries
  const int wgBase = blockIdx.x * 256;

  // ---- stage encoder weights (f32 -> f16, zero-padded) --------------------
  for (int f = tid; f < 64 * 64; f += 256) {
    int n = f >> 6, k = f & 63;
    sW1[f] = (_Float16)((k < 39) ? enc_w1[n * 39 + k] : 0.0f);
  }
  for (int f = tid; f < 32 * 64; f += 256)
    sW2[f] = (_Float16)enc_w2[f];
  for (int f = tid; f < 16 * 32; f += 256) {
    int n = f >> 5, k = f & 31;
    sW3[f] = (_Float16)((n < 4) ? enc_w3[n * 32 + k] : 0.0f);
  }
  // ---- stage 256 state rows (coalesced f32 loads, f16 in LDS, K padded) ---
  for (int f = tid; f < 256 * 64; f += 256) {
    int row = f >> 6, k = f & 63;
    float v = (k < 39) ? state[(size_t)(wgBase + row) * 39 + k] : 0.0f;
    sX[row >> 5][(row & 31) * 64 + k] = (_Float16)v;
  }
  __syncthreads();

  _Float16* Xw = sX[wave];

  // ================= encoder: WMMA, two M=16 tiles per wave ================
  #pragma unroll
  for (int m = 0; m < 2; ++m) {
    const _Float16* arow = Xw + (m * 16 + laneN) * 64;

    // ---- layer 1: [16x64] x [64x64] -> relu, 8 WMMA ----
    v16h a0 = load_a_frag(arow, 0 + laneH * 16);
    v16h a1 = load_a_frag(arow, 32 + laneH * 16);
    #pragma unroll
    for (int n = 0; n < 4; ++n) {
      const _Float16* brow = sW1 + (n * 16 + laneN) * 64;
      v16h b0 = *(const v16h*)(brow + laneH * 16);
      v16h b1 = *(const v16h*)(brow + 32 + laneH * 16);
      v8f c = {};
      c = wmma_f16(a0, b0, c);
      c = wmma_f16(a1, b1, c);
      float bias = enc_b1[n * 16 + laneN];
      #pragma unroll
      for (int r = 0; r < 8; ++r) {
        float h = c[r] + bias;
        h = h > 0.0f ? h : 0.0f;
        // D layout: row = r + 8*laneH, col = laneN (within this N-tile)
        Xw[(m * 16 + r + 8 * laneH) * 64 + n * 16 + laneN] = (_Float16)h;
      }
    }

    // ---- layer 2: [16x64] x [64x32] -> relu, 4 WMMA ----
    v16h a20 = load_a_frag(arow, 0 + laneH * 16);
    v16h a21 = load_a_frag(arow, 32 + laneH * 16);
    #pragma unroll
    for (int n = 0; n < 2; ++n) {
      const _Float16* brow = sW2 + (n * 16 + laneN) * 64;
      v16h b0 = *(const v16h*)(brow + laneH * 16);
      v16h b1 = *(const v16h*)(brow + 32 + laneH * 16);
      v8f c = {};
      c = wmma_f16(a20, b0, c);
      c = wmma_f16(a21, b1, c);
      float bias = enc_b2[n * 16 + laneN];
      #pragma unroll
      for (int r = 0; r < 8; ++r) {
        float h = c[r] + bias;
        h = h > 0.0f ? h : 0.0f;
        Xw[(m * 16 + r + 8 * laneH) * 64 + n * 16 + laneN] = (_Float16)h;
      }
    }

    // ---- layer 3: [16x32] x [32x16(pad of 4)] -> tanh, 1 WMMA ----
    {
      v16h a3 = load_a_frag(arow, laneH * 16);
      const _Float16* brow = sW3 + laneN * 32;
      v16h b = *(const v16h*)(brow + laneH * 16);
      v8f c = {};
      c = wmma_f16(a3, b, c);
      if (laneN < 4) {
        float bias = enc_b3[laneN];
        #pragma unroll
        for (int r = 0; r < 8; ++r) {
          float e = tanhf(c[r] + bias);
          sEnc[wave][(m * 16 + r + 8 * laneH) * 4 + laneN] = e;
        }
      }
    }
  }

  // ============ per-lane: 4-qubit statevector sim + decoder ================
  // Same-wave LDS producer/consumer: hardware keeps LDS ops in order.
  float enc4[4];
  #pragma unroll
  for (int i = 0; i < 4; ++i) enc4[i] = sEnc[wave][lane * 4 + i];

  float qp[8];  // uniform -> scalar loads / K$
  #pragma unroll
  for (int i = 0; i < 8; ++i) qp[i] = qparams[i];

  // single-qubit amplitudes: amp0 = cos(th/2) e^{-i ph/2}, amp1 = sin(th/2) e^{+i ph/2}
  float A0r[4], A0i[4], A1r[4], A1i[4];
  #pragma unroll
  for (int i = 0; i < 4; ++i) {
    float th = 0.5f * (enc4[i] * PI_F + qp[2 * i]);
    float ph = 0.5f * qp[2 * i + 1];
    float sh, ch, sp, cp;
    __sincosf(th, &sh, &ch);
    __sincosf(ph, &sp, &cp);
    A0r[i] = ch * cp; A0i[i] = -ch * sp;
    A1r[i] = sh * cp; A1i[i] = sh * sp;
  }

  // psi = q0 (x) q1 (x) q2 (x) q3 ; state index s = i*8 + j*4 + k*2 + l
  float wr[16], wi[16];
  wr[0] = A0r[0]; wi[0] = A0i[0];
  wr[1] = A1r[0]; wi[1] = A1i[0];
  #pragma unroll
  for (int s = 1; s >= 0; --s) {           // append qubit 1
    float xr = wr[s], xi = wi[s];
    wr[2*s]   = xr * A0r[1] - xi * A0i[1]; wi[2*s]   = xr * A0i[1] + xi * A0r[1];
    wr[2*s+1] = xr * A1r[1] - xi * A1i[1]; wi[2*s+1] = xr * A1i[1] + xi * A1r[1];
  }
  #pragma unroll
  for (int s = 3; s >= 0; --s) {           // append qubit 2
    float xr = wr[s], xi = wi[s];
    wr[2*s]   = xr * A0r[2] - xi * A0i[2]; wi[2*s]   = xr * A0i[2] + xi * A0r[2];
    wr[2*s+1] = xr * A1r[2] - xi * A1i[2]; wi[2*s+1] = xr * A1i[2] + xi * A1r[2];
  }
  #pragma unroll
  for (int s = 7; s >= 0; --s) {           // append qubit 3
    float xr = wr[s], xi = wi[s];
    wr[2*s]   = xr * A0r[3] - xi * A0i[3]; wi[2*s]   = xr * A0i[3] + xi * A0r[3];
    wr[2*s+1] = xr * A1r[3] - xi * A1i[3]; wi[2*s+1] = xr * A1i[3] + xi * A1r[3];
  }

  // CNOT(0,1): control bit3 flips bit2 ; CNOT(1,2): bit2->bit1 ; CNOT(2,3): bit1->bit0
  #pragma unroll
  for (int s = 0; s < 16; ++s)
    if ((s & 8) && !(s & 4)) {
      float t = wr[s]; wr[s] = wr[s ^ 4]; wr[s ^ 4] = t;
      t = wi[s]; wi[s] = wi[s ^ 4]; wi[s ^ 4] = t;
    }
  #pragma unroll
  for (int s = 0; s < 16; ++s)
    if ((s & 4) && !(s & 2)) {
      float t = wr[s]; wr[s] = wr[s ^ 2]; wr[s ^ 2] = t;
      t = wi[s]; wi[s] = wi[s ^ 2]; wi[s ^ 2] = t;
    }
  #pragma unroll
  for (int s = 0; s < 16; ++s)
    if ((s & 2) && !(s & 1)) {
      float t = wr[s]; wr[s] = wr[s ^ 1]; wr[s ^ 1] = t;
      t = wi[s]; wi[s] = wi[s ^ 1]; wi[s ^ 1] = t;
    }

  // probabilities -> <Z_q>
  float ev[4] = {0.f, 0.f, 0.f, 0.f};
  #pragma unroll
  for (int s = 0; s < 16; ++s) {
    float p = wr[s] * wr[s] + wi[s] * wi[s];
    ev[0] += (s & 8) ? -p : p;
    ev[1] += (s & 4) ? -p : p;
    ev[2] += (s & 2) ? -p : p;
    ev[3] += (s & 1) ? -p : p;
  }

  // ---- decoder: weights are uniform -> scalar/K$ loads, FMAs on VALU ----
  float d1[32];
  #pragma unroll
  for (int o = 0; o < 32; ++o) {
    float t = dec_b1[o];
    #pragma unroll
    for (int i = 0; i < 4; ++i) t = fmaf(dec_w1[o * 4 + i], ev[i], t);
    d1[o] = t > 0.0f ? t : 0.0f;
  }
  float d2[16];
  #pragma unroll
  for (int o = 0; o < 16; ++o) {
    float t = dec_b2[o];
    #pragma unroll
    for (int i = 0; i < 32; ++i) t = fmaf(dec_w2[o * 32 + i], d1[i], t);
    d2[o] = t > 0.0f ? t : 0.0f;
  }
  float res[8];
  #pragma unroll
  for (int o = 0; o < 8; ++o) {
    float t = dec_b3[o];
    #pragma unroll
    for (int i = 0; i < 16; ++i) t = fmaf(dec_w3[o * 16 + i], d2[i], t);
    res[o] = t;
  }

  const size_t sample = (size_t)wgBase + (size_t)wave * 32 + lane;
  float4 lo4 = make_float4(res[0], res[1], res[2], res[3]);
  float4 hi4 = make_float4(res[4], res[5], res[6], res[7]);
  *(float4*)(out + sample * 8)     = lo4;
  *(float4*)(out + sample * 8 + 4) = hi4;
}

extern "C" void kernel_launch(void* const* d_in, const int* in_sizes, int n_in,
                              void* d_out, int out_size, void* d_ws, size_t ws_size,
                              hipStream_t stream) {
  (void)n_in; (void)out_size; (void)d_ws; (void)ws_size;
  const float* state  = (const float*)d_in[0];
  const float* enc_w1 = (const float*)d_in[1];
  const float* enc_b1 = (const float*)d_in[2];
  const float* enc_w2 = (const float*)d_in[3];
  const float* enc_b2 = (const float*)d_in[4];
  const float* enc_w3 = (const float*)d_in[5];
  const float* enc_b3 = (const float*)d_in[6];
  const float* qpar   = (const float*)d_in[7];
  const float* dec_w1 = (const float*)d_in[8];
  const float* dec_b1 = (const float*)d_in[9];
  const float* dec_w2 = (const float*)d_in[10];
  const float* dec_b2 = (const float*)d_in[11];
  const float* dec_w3 = (const float*)d_in[12];
  const float* dec_b3 = (const float*)d_in[13];

  const int B = in_sizes[0] / 39;          // 262144
  const int grid = B / 256;                // 256 samples per block
  qai_fused_kernel<<<dim3(grid), dim3(256), 0, stream>>>(
      state, enc_w1, enc_b1, enc_w2, enc_b2, enc_w3, enc_b3, qpar,
      dec_w1, dec_b1, dec_w2, dec_b2, dec_w3, dec_b3, (float*)d_out);
}
